// RegionGroupPooling_3865470566552
// MI455X (gfx1250) — matrix operand order repaired and checked by reading, not compiled, not run on hardware
//
#include <hip/hip_runtime.h>
#include <hip/hip_bf16.h>
#include <math.h>

typedef __attribute__((ext_vector_type(16))) _Float16 v16h;
typedef __attribute__((ext_vector_type(8)))  float    v8f;

#define BDIM  32
#define CDIM  512
#define HWDIM 4096
#define TOPK  32

// -------------------------------------------------------------------------
// Kernel 1: scores[b,n] = sum_c X[b,c,n] * W[c]
// One wave computes 16 locations' scores via chained v_wmma_f32_16x16x32_f16.
//   A (16x32 f16): every row = W chunk (rows 1..15 feed D rows we ignore,
//                  and the A K-pattern is row-independent, so no masking)
//   B (32x16 f16): K = channels, N = locations (lane-striped, coalesced)
//   D[0,N] (f32)  = scores for the 16 locations (lanes 0..15, VGPR0)
// -------------------------------------------------------------------------
__global__ __launch_bounds__(256) void rgp_score_kernel(
    const float* __restrict__ X, const float* __restrict__ W,
    float* __restrict__ scores) {
  const int wave  = blockIdx.x * (blockDim.x >> 5) + (threadIdx.x >> 5);
  const int lane  = threadIdx.x & 31;
  const int tilesPerBatch = HWDIM / 16;               // 256
  const int b     = wave / tilesPerBatch;
  const int nbase = (wave % tilesPerBatch) * 16;
  const int m     = lane & 15;                        // col N for B / D
  const int hi    = lane >> 4;                        // half-wave select
  const float* __restrict__ Xb = X + (size_t)b * CDIM * HWDIM;

  v8f acc = {};
#pragma unroll 4
  for (int cbase = 0; cbase < CDIM; cbase += 32) {
    // ---- A fragment: unconditional W loads (row-independent K pattern).
    //      Elements (2v, 2v+1) are consecutive channels -> float2 loads.
    v16h a;
#pragma unroll
    for (int v = 0; v < 8; ++v) {
      const int k = 2 * v + hi * 16;
      const float2 wp = *(const float2*)(W + cbase + k);   // 8B aligned
      a[2 * v]     = (_Float16)wp.x;
      a[2 * v + 1] = (_Float16)wp.y;
    }
    // ---- B fragment: 16-bit B 32x16 layout; channel rows strided by HWDIM
    v16h bm;
#pragma unroll
    for (int e = 0; e < 16; ++e) {
      const int v = e >> 1, h = e & 1;
      const int k = hi * 16 + 2 * v + h;              // channel within tile
      const float xv = Xb[(size_t)(cbase + k) * HWDIM + (nbase + m)];
      bm[e] = (_Float16)xv;
    }
    acc = __builtin_amdgcn_wmma_f32_16x16x32_f16(
        /*neg_a=*/false, a, /*neg_b=*/false, bm,
        /*c_mod=*/(short)0, acc, /*reuse_a=*/false, /*reuse_b=*/false);
  }
  // D row 0: VGPR0, lanes 0..15 hold N = 0..15
  if (lane < 16) scores[b * HWDIM + nbase + lane] = acc[0];
}

// -------------------------------------------------------------------------
// Kernel 2: per-batch top-32 indices (iterative LDS argmax, tie -> low idx)
// -------------------------------------------------------------------------
__global__ __launch_bounds__(256) void rgp_topk_kernel(
    const float* __restrict__ scores, int* __restrict__ idxs) {
  __shared__ float s[HWDIM];
  __shared__ float rv[256];
  __shared__ int   ri[256];
  const int b = blockIdx.x;
  const int t = threadIdx.x;
  for (int i = t; i < HWDIM; i += 256) s[i] = scores[b * HWDIM + i];
  __syncthreads();

  for (int k = 0; k < TOPK; ++k) {
    float best = -INFINITY;
    int   bi   = 0;
#pragma unroll
    for (int j = 0; j < HWDIM / 256; ++j) {
      const int   i = t * (HWDIM / 256) + j;          // ascending: low-idx tie
      const float v = s[i];
      if (v > best) { best = v; bi = i; }
    }
    rv[t] = best; ri[t] = bi;
    __syncthreads();
    for (int stride = 128; stride > 0; stride >>= 1) {
      if (t < stride) {
        const float ov = rv[t + stride];
        const int   oi = ri[t + stride];
        if (ov > rv[t] || (ov == rv[t] && oi < ri[t])) { rv[t] = ov; ri[t] = oi; }
      }
      __syncthreads();
    }
    if (t == 0) { idxs[b * TOPK + k] = ri[0]; s[ri[0]] = -INFINITY; }
    __syncthreads();
  }
}

// -------------------------------------------------------------------------
// Kernel 3: feat[b,c] = mean_k X[b,c,idx[b,k]] (exact fp32);
//           logit[b]  = feat[b,:]·W + bias  (LDS tree reduction)
// -------------------------------------------------------------------------
__global__ __launch_bounds__(512) void rgp_feat_kernel(
    const float* __restrict__ X, const float* __restrict__ W,
    const float* __restrict__ bias, const int* __restrict__ idxs,
    float* __restrict__ out) {
  __shared__ int   sid[TOPK];
  __shared__ float red[512];
  const int b = blockIdx.x;
  const int c = threadIdx.x;
  if (c < TOPK) sid[c] = idxs[b * TOPK + c];
  __syncthreads();

  const float* __restrict__ Xbc =
      X + (size_t)b * CDIM * HWDIM + (size_t)c * HWDIM;
  float acc = 0.0f;
#pragma unroll
  for (int k = 0; k < TOPK; ++k) acc += Xbc[sid[k]];
  const float feat = acc * (1.0f / (float)TOPK);
  out[b * CDIM + c] = feat;                            // feat block of d_out

  red[c] = feat * W[c];
  __syncthreads();
  for (int s = 256; s > 0; s >>= 1) {
    if (c < s) red[c] += red[c + s];
    __syncthreads();
  }
  if (c == 0) out[BDIM * CDIM + b] = red[0] + bias[0]; // logit block of d_out
}

// -------------------------------------------------------------------------
extern "C" void kernel_launch(void* const* d_in, const int* in_sizes, int n_in,
                              void* d_out, int out_size, void* d_ws,
                              size_t ws_size, hipStream_t stream) {
  (void)in_sizes; (void)n_in; (void)out_size; (void)ws_size;
  const float* X    = (const float*)d_in[0];
  const float* W    = (const float*)d_in[1];
  const float* bias = (const float*)d_in[2];
  float* out = (float*)d_out;

  float* scores = (float*)d_ws;                                   // 512 KB
  int*   idxs   = (int*)((char*)d_ws +
                         (size_t)BDIM * HWDIM * sizeof(float));   // 4 KB

  // 8192 wave-tiles (16 locations each), 8 waves per 256-thread block
  rgp_score_kernel<<<BDIM * (HWDIM / 16) / 8, 256, 0, stream>>>(X, W, scores);
  rgp_topk_kernel<<<BDIM, 256, 0, stream>>>(scores, idxs);
  rgp_feat_kernel<<<BDIM, 512, 0, stream>>>(X, W, bias, idxs, out);
}
